// EFN_10943576670835
// MI455X (gfx1250) — compile-verified, matrix operand order checked by reading, not run on hardware
//
#include <hip/hip_runtime.h>
#include <hip/hip_fp16.h>

typedef _Float16 v16h  __attribute__((ext_vector_type(16)));
typedef _Float16 half8 __attribute__((ext_vector_type(8)));
typedef float    v8f   __attribute__((ext_vector_type(8)));

#define HIDDEN 128
#define WPB 2                 // waves per block
#define BLOCK (WPB * 32)

// Convert W2 (f32, [128 in,128 out] row-major) to W2^T in f16: w2t[n*128+k] = W2[k,n]
__global__ void w2_transpose_f16(const float* __restrict__ W2, _Float16* __restrict__ w2t) {
  int i = blockIdx.x * blockDim.x + threadIdx.x;     // i = n*128 + k
  if (i < HIDDEN * HIDDEN) {
    int n = i >> 7, k = i & 127;
    w2t[i] = (_Float16)W2[k * HIDDEN + n];
  }
}

__global__ __launch_bounds__(BLOCK) void ptconv_fused(
    const float* __restrict__ x, const float* __restrict__ pos,
    const float* __restrict__ W1, const float* __restrict__ b1,
    const float* __restrict__ b2, const _Float16* __restrict__ w2t,
    const int* __restrict__ srcIdx, const int* __restrict__ dstIdx,
    float* __restrict__ out, float* __restrict__ dump, int nEdges)
{
  __shared__ _Float16 bts[HIDDEN * HIDDEN];       // W2^T f16 (32 KB)
  __shared__ float    w1s[3 * HIDDEN];
  __shared__ float    b1s[HIDDEN];
  __shared__ float    b2s[HIDDEN];
  __shared__ _Float16 hbuf[WPB][32 * HIDDEN];     // per-wave 32x128 f16 A tiles (8 KB)

  const int tid = threadIdx.x;
  for (int i = tid; i < HIDDEN * HIDDEN; i += BLOCK) bts[i] = w2t[i];
  for (int i = tid; i < 3 * HIDDEN; i += BLOCK)      w1s[i] = W1[i];
  for (int i = tid; i < HIDDEN; i += BLOCK) { b1s[i] = b1[i]; b2s[i] = b2[i]; }
  __syncthreads();

  const int lane = tid & 31;
  const int wave = tid >> 5;
  const int m    = lane & 15;    // row within a 16-edge subtile / col within an N-tile
  const int hsel = lane >> 4;    // which K-half of a fragment this lane owns
  _Float16* hrow0 = &hbuf[wave][m * HIDDEN];            // subtile 0, edge m
  _Float16* hrow1 = &hbuf[wave][(16 + m) * HIDDEN];     // subtile 1, edge 16+m

  // loop-invariant per lane: bias for the 8 N-tiles, and scatter column base
  float bias8[8];
#pragma unroll
  for (int nt = 0; nt < 8; ++nt) bias8[nt] = b2s[nt * 16 + m];
  float* const outm  = out  + m;
  float* const dumpm = dump + m;

  const int nChunks     = (nEdges + 31) >> 5;           // 32 edges per wave iteration
  const int chunkStride = gridDim.x * WPB;

  for (int chunk = blockIdx.x * WPB + wave; chunk < nChunks; chunk += chunkStride) {
    // ---- prefetch next chunk's edge indices (the only HBM-streamed input) ----
    {
      int eN = (chunk + chunkStride) * 32 + lane;
      eN = eN < nEdges ? eN : nEdges - 1;
      __builtin_prefetch(&srcIdx[eN], 0, 0);
      __builtin_prefetch(&dstIdx[eN], 0, 0);
    }

    // ---- gather per-edge inputs: one edge per lane (32 edges; L2-resident) ----
    float xj = 0.f, dx = 0.f, dy = 0.f;
    int dsti = -1;
    {
      int e = (chunk << 5) + lane;
      if (e < nEdges) {
        int s = srcIdx[e];
        int d = dstIdx[e];
        xj   = x[s];
        dx   = pos[2 * s]     - pos[2 * d];
        dy   = pos[2 * s + 1] - pos[2 * d + 1];
        dsti = d;
      }
    }
    const float xs0 = __shfl(xj, m, 32),      dxs0 = __shfl(dx, m, 32),      dys0 = __shfl(dy, m, 32);
    const float xs1 = __shfl(xj, 16 + m, 32), dxs1 = __shfl(dx, 16 + m, 32), dys1 = __shfl(dy, 16 + m, 32);

    // ---- scatter base pointers for the 16 C/D rows this lane will hold ----
    // address of atomic r at N-tile nt is pa[r] + nt*16 floats -> constant byte
    // offset nt*64 folds into the global_atomic instruction offset field.
    float* pa[8];
    float* pb[8];
#pragma unroll
    for (int r = 0; r < 8; ++r) {
      int da = __shfl(dsti, r + hsel * 8, 32);
      int db = __shfl(dsti, 16 + r + hsel * 8, 32);
      pa[r] = (da >= 0) ? (outm + (long)da * HIDDEN) : dumpm;
      pb[r] = (db >= 0) ? (outm + (long)db * HIDDEN) : dumpm;
    }

    // ---- layer 1 (3->128) + ReLU for both subtiles; shared W1 broadcasts ----
    const int kb = hsel * 64;
#pragma unroll
    for (int kk = 0; kk < 64; kk += 8) {
      half8 h0, h1;
#pragma unroll
      for (int j = 0; j < 8; ++j) {
        int k = kb + kk + j;
        float w0 = w1s[k], w1 = w1s[HIDDEN + k], w2 = w1s[2 * HIDDEN + k], bb = b1s[k];
        float a0 = fmaf(xs0, w0, fmaf(dxs0, w1, fmaf(dys0, w2, bb)));
        float a1 = fmaf(xs1, w0, fmaf(dxs1, w1, fmaf(dys1, w2, bb)));
        h0[j] = (_Float16)fmaxf(a0, 0.f);
        h1[j] = (_Float16)fmaxf(a1, 0.f);
      }
      *(half8*)(&hrow0[kb + kk]) = h0;              // ds_store_b128
      *(half8*)(&hrow1[kb + kk]) = h1;
    }
    __builtin_amdgcn_wave_barrier();                // DS pipe is in-order per wave

    // ---- A fragments: 2 subtiles x 4 K-chunks (ISA 16-bit A layout) ----
    v16h af[2][4];
#pragma unroll
    for (int s = 0; s < 2; ++s) {
#pragma unroll
      for (int kc = 0; kc < 4; ++kc) {
        const _Float16* ap = &hbuf[wave][(s * 16 + m) * HIDDEN + kc * 32 + hsel * 8];
        half8 lo = *(const half8*)ap;               // K: base..base+7
        half8 hi = *(const half8*)(ap + 16);        // K: base+16..base+23
        af[s][kc] = __builtin_shufflevector(lo, hi, 0,1,2,3,4,5,6,7,8,9,10,11,12,13,14,15);
      }
    }

    // ---- layer 2: two [16x128] @ [128x128]; each B fragment feeds 2 WMMAs ----
#pragma unroll
    for (int nt = 0; nt < 8; ++nt) {
      const int n = nt * 16 + m;
      const float bias = bias8[nt];
      v8f acc0 = { bias, bias, bias, bias, bias, bias, bias, bias };
      v8f acc1 = acc0;
#pragma unroll
      for (int kc = 0; kc < 4; ++kc) {
        const _Float16* bp = &bts[n * HIDDEN + kc * 32 + hsel * 8];
        half8 lo = *(const half8*)bp;
        half8 hi = *(const half8*)(bp + 16);
        v16h bfrag = __builtin_shufflevector(lo, hi, 0,1,2,3,4,5,6,7,8,9,10,11,12,13,14,15);
        acc0 = __builtin_amdgcn_wmma_f32_16x16x32_f16(false, af[0][kc], false, bfrag,
                                                      (short)0, acc0, false, false);
        acc1 = __builtin_amdgcn_wmma_f32_16x16x32_f16(false, af[1][kc], false, bfrag,
                                                      (short)0, acc1, false, false);
      }
      // pure-VMEM scatter: base pointer + constant instruction offset (nt*64 B)
#pragma unroll
      for (int r = 0; r < 8; ++r) {
        __hip_atomic_fetch_add(pa[r] + nt * 16, acc0[r],
                               __ATOMIC_RELAXED, __HIP_MEMORY_SCOPE_AGENT);
        __hip_atomic_fetch_add(pb[r] + nt * 16, acc1[r],
                               __ATOMIC_RELAXED, __HIP_MEMORY_SCOPE_AGENT);
      }
    }
  }
}

extern "C" void kernel_launch(void* const* d_in, const int* in_sizes, int n_in,
                              void* d_out, int out_size, void* d_ws, size_t ws_size,
                              hipStream_t stream) {
  const float* x   = (const float*)d_in[0];
  const float* pos = (const float*)d_in[1];
  const float* W1  = (const float*)d_in[2];
  const float* b1  = (const float*)d_in[3];
  const float* W2  = (const float*)d_in[4];
  const float* b2  = (const float*)d_in[5];
  const int*   ei  = (const int*)d_in[6];

  const int nEdges  = in_sizes[6] / 2;
  const int* srcIdx = ei;            // edge_index[0,:]
  const int* dstIdx = ei + nEdges;   // edge_index[1,:]

  _Float16* w2t  = (_Float16*)d_ws;                       // 32 KB: W2^T f16
  float*    dump = (float*)((char*)d_ws + 32768);         // 512 B: scatter sink

  hipMemsetAsync(d_out, 0, (size_t)out_size * sizeof(float), stream);
  w2_transpose_f16<<<(HIDDEN * HIDDEN + 255) / 256, 256, 0, stream>>>(W2, w2t);

  const int nChunks = (nEdges + 31) / 32;
  int blocks = (nChunks + WPB - 1) / WPB;
  if (blocks > 3072) blocks = 3072;  // persistent waves, grid-stride over 32-edge chunks
  ptconv_fused<<<blocks, BLOCK, 0, stream>>>(x, pos, W1, b1, b2, w2t,
                                             srcIdx, dstIdx, (float*)d_out, dump, nEdges);
}
